// trunk_model_80006650789995
// MI455X (gfx1250) — matrix operand order repaired and checked by reading, not compiled, not run on hardware
//
#include <hip/hip_runtime.h>
#include <math.h>

typedef __bf16 bf16;
typedef __attribute__((ext_vector_type(4)))  bf16  v4bf;
typedef __attribute__((ext_vector_type(8)))  bf16  v8bf;
typedef __attribute__((ext_vector_type(16))) bf16  v16bf;
typedef __attribute__((ext_vector_type(4)))  float v4f;
typedef __attribute__((ext_vector_type(8)))  float v8f;

#define SEQ 1024
#define DM  768
#define NH  12
#define HD  64
#define FF  3072
#define NL  12
#define VOC 50257

__device__ __forceinline__ float gelu_new(float x) {
  return 0.5f * x * (1.0f + tanhf(0.7978845608028654f * (x + 0.044715f * x * x * x)));
}

// ---------------- embedding: x[s,:] = wte[ids[s],:] + wpe[s,:] ----------------
__global__ void embed_kernel(const int* __restrict__ ids,
                             const float* __restrict__ wte,
                             const float* __restrict__ wpe,
                             float* __restrict__ x) {
  const int s = blockIdx.x;
  const int id = ids[s];
  for (int d = threadIdx.x; d < DM; d += blockDim.x)
    x[s * DM + d] = wte[(size_t)id * DM + d] + wpe[s * DM + d];
}

// ---------------- layernorm: fp32 in -> bf16 out ----------------
__global__ void ln_kernel(const float* __restrict__ x,
                          const float* __restrict__ g,
                          const float* __restrict__ b,
                          bf16* __restrict__ y) {
  __shared__ float r1[256], r2[256];
  const int s = blockIdx.x;
  const float* xr = x + s * DM;
  float s1 = 0.f, s2 = 0.f;
  for (int d = threadIdx.x; d < DM; d += 256) {
    float v = xr[d];
    s1 += v; s2 += v * v;
  }
  r1[threadIdx.x] = s1; r2[threadIdx.x] = s2;
  __syncthreads();
  for (int off = 128; off > 0; off >>= 1) {
    if (threadIdx.x < off) {
      r1[threadIdx.x] += r1[threadIdx.x + off];
      r2[threadIdx.x] += r2[threadIdx.x + off];
    }
    __syncthreads();
  }
  const float mu  = r1[0] * (1.0f / DM);
  const float var = r2[0] * (1.0f / DM) - mu * mu;
  const float inv = rsqrtf(var + 1e-5f);
  for (int d = threadIdx.x; d < DM; d += 256)
    y[s * DM + d] = (bf16)((xr[d] - mu) * inv * g[d] + b[d]);
}

// ------------- causal softmax in-place on bf16 scores, scale 1/8 -------------
__global__ void softmax_kernel(bf16* __restrict__ sc) {
  __shared__ float red[256];
  const int s = blockIdx.x, h = blockIdx.y;
  bf16* row = sc + (h * SEQ + s) * SEQ;
  const int len = s + 1;
  float lmax = -3.0e38f;
  for (int t = threadIdx.x; t < len; t += 256)
    lmax = fmaxf(lmax, (float)row[t] * 0.125f);
  red[threadIdx.x] = lmax;
  __syncthreads();
  for (int off = 128; off > 0; off >>= 1) {
    if (threadIdx.x < off) red[threadIdx.x] = fmaxf(red[threadIdx.x], red[threadIdx.x + off]);
    __syncthreads();
  }
  const float rmax = red[0];
  __syncthreads();
  float lsum = 0.f;
  for (int t = threadIdx.x; t < len; t += 256)
    lsum += __expf((float)row[t] * 0.125f - rmax);
  red[threadIdx.x] = lsum;
  __syncthreads();
  for (int off = 128; off > 0; off >>= 1) {
    if (threadIdx.x < off) red[threadIdx.x] += red[threadIdx.x + off];
    __syncthreads();
  }
  const float inv = 1.0f / red[0];
  for (int t = threadIdx.x; t < SEQ; t += 256) {
    float p = (t < len) ? __expf((float)row[t] * 0.125f - rmax) * inv : 0.0f;
    row[t] = (bf16)p;  // zero-padded past diagonal -> attn@V can run full K
  }
}

// ---------------- bf16 WMMA GEMM ----------------
// Block tile: (128*MT) x 64, K step 32.  8 waves; each wave owns MT 16-row
// strips and 4 accumulator tiles per strip (B tile reused MT times).
// B modes (LDB = compile-time leading stride of B source):
//   0: fp32, n-contiguous rows  (weights W[k][n], stride LDB along k)
//   1: bf16, k-contiguous       (K^T: B[k][n] = src[n*LDB + k]) -> async to LDS
//   2: bf16, n-contiguous       (V:   B[k][n] = src[k*LDB + n])
//   3: fp32, k-contiguous, N-guarded (wte^T logits)
#define LDSB_STRIDE 40  // 32 K-elems + pad (40*2B = 80B = 5*16B, keeps alignment)

template<int BMODE, int LDB, int MT, bool BIAS, bool RESID, bool GELU_, bool STF32, bool STBF16>
__global__ __launch_bounds__(256)
void gemm_wmma(const bf16* __restrict__ A, int lda, int strideAz,
               const void* __restrict__ Bv, int strideBz,
               const float* __restrict__ bias,
               const float* __restrict__ resid, int ldr,
               float* __restrict__ Cf, int ldc, int strideCfz,
               bf16* __restrict__ Cb, int ldcb, int strideCbz,
               int M, int N, int K) {
  constexpr bool NGUARD = (BMODE == 3);
  constexpr bool BF16B  = (BMODE == 1) || (BMODE == 2);
  __shared__ bf16 ldsB[64 * LDSB_STRIDE];

  const int z = blockIdx.z;
  A += z * strideAz;
  const float* Bf = (const float*)Bv + (BF16B ? 0 : z * strideBz);
  const bf16*  Bb = (const bf16*)Bv  + (BF16B ? z * strideBz : 0);
  if (STF32)  Cf += z * strideCfz;
  if (STBF16) Cb += z * strideCbz;

  const int tid  = threadIdx.x;
  const int wave = tid >> 5;
  const int lane = tid & 31;
  const int hi16 = lane >> 4;   // half-wave -> K sub-range select
  const int l16  = lane & 15;

  const int nBase = blockIdx.x * 64;
  const int m0    = blockIdx.y * (128 * MT) + wave * 16;

  v8f acc[MT][4] = {};

  const bf16* Arow[MT];
  #pragma unroll
  for (int mt = 0; mt < MT; ++mt)
    Arow[mt] = A + (m0 + mt * 128 + l16) * lda + 8 * hi16;

  for (int k0 = 0; k0 < K; k0 += 32) {
    __syncthreads();
    // ---- stage B tile [32 x 64] -> LDS transposed [n][k] ----
    if (BMODE == 0) {            // fp32 weights, n-contiguous
      const int kp = (k0 + 32 < K) ? k0 + 32 : k0;  // branch-free prefetch clamp
      #pragma unroll
      for (int it = 0; it < 2; ++it) {
        const int v  = tid + it * 256;   // 0..511 vec4 units
        const int n  = v & 63;
        const int k4 = v >> 6;           // 0..7
        const float* bp = Bf + (k0 + k4 * 4) * LDB + nBase + n;
        __builtin_prefetch(Bf + (kp + k4 * 4) * LDB + nBase + n, 0, 3);
        v4bf hv;
        hv[0] = (bf16)bp[0];
        hv[1] = (bf16)bp[LDB];
        hv[2] = (bf16)bp[2 * LDB];
        hv[3] = (bf16)bp[3 * LDB];
        *(v4bf*)&ldsB[n * LDSB_STRIDE + k4 * 4] = hv;
      }
    } else if (BMODE == 1) {     // bf16, k-contiguous: async DMA global->LDS
      const int n  = tid >> 2;           // 0..63
      const int k8 = tid & 3;            // 0..3
      const bf16* gp = Bb + (nBase + n) * LDB + k0 + k8 * 8;
      const unsigned lof = (unsigned)(size_t)&ldsB[n * LDSB_STRIDE + k8 * 8];
      asm volatile("global_load_async_to_lds_b128 %0, %1, off"
                   :: "v"(lof), "v"(gp) : "memory");
    } else if (BMODE == 2) {     // bf16, n-contiguous
      #pragma unroll
      for (int it = 0; it < 2; ++it) {
        const int v  = tid + it * 256;
        const int n  = v & 63;
        const int k4 = v >> 6;
        const bf16* bp = Bb + (k0 + k4 * 4) * LDB + nBase + n;
        v4bf hv;
        hv[0] = bp[0];
        hv[1] = bp[LDB];
        hv[2] = bp[2 * LDB];
        hv[3] = bp[3 * LDB];
        *(v4bf*)&ldsB[n * LDSB_STRIDE + k4 * 4] = hv;
      }
    } else {                     // fp32, k-contiguous, N-guarded (wte^T)
      #pragma unroll
      for (int it = 0; it < 2; ++it) {
        const int v  = tid + it * 256;
        const int n  = v >> 3;           // 0..63
        const int k4 = v & 7;            // 0..7
        const bf16 z0 = (bf16)0.0f;
        v4bf hv = {z0, z0, z0, z0};
        if (nBase + n < N) {
          const v4f f = *(const v4f*)(Bf + (nBase + n) * LDB + k0 + k4 * 4);
          hv[0] = (bf16)f[0]; hv[1] = (bf16)f[1];
          hv[2] = (bf16)f[2]; hv[3] = (bf16)f[3];
        }
        *(v4bf*)&ldsB[n * LDSB_STRIDE + k4 * 4] = hv;
      }
    }
    if (BMODE == 1)
      asm volatile("s_wait_asynccnt 0x0" ::: "memory");  // LDS writes visible pre-barrier
    __syncthreads();

    // ---- A fragments: 16-bit A 16x32 layout: e -> K = (e<8?e:e+8) + 8*hi16 ----
    v16bf afrag[MT];
    #pragma unroll
    for (int mt = 0; mt < MT; ++mt) {
      const v8bf a_lo = *(const v8bf*)(Arow[mt] + k0);
      const v8bf a_hi = *(const v8bf*)(Arow[mt] + k0 + 16);
      #pragma unroll
      for (int i = 0; i < 8; ++i) { afrag[mt][i] = a_lo[i]; afrag[mt][i + 8] = a_hi[i]; }
    }

    #pragma unroll
    for (int nt = 0; nt < 4; ++nt) {
      // B fragment: e -> K = (e<8?e:e+8) + 8*hi16, N = l16 (matches A layout)
      const bf16* bp = &ldsB[(nt * 16 + l16) * LDSB_STRIDE + 8 * hi16];
      const v8bf b_lo = *(const v8bf*)(bp);
      const v8bf b_hi = *(const v8bf*)(bp + 16);
      v16bf bfrag;
      #pragma unroll
      for (int i = 0; i < 8; ++i) { bfrag[i] = b_lo[i]; bfrag[i + 8] = b_hi[i]; }
      #pragma unroll
      for (int mt = 0; mt < MT; ++mt)
        acc[mt][nt] = __builtin_amdgcn_wmma_f32_16x16x32_bf16(
            false, afrag[mt], false, bfrag, (short)0, acc[mt][nt], false, false);
    }
  }

  // ---- epilogue: C/D layout: VGPR r -> M = r + 8*hi16, N = l16 ----
  #pragma unroll
  for (int mt = 0; mt < MT; ++mt) {
    #pragma unroll
    for (int nt = 0; nt < 4; ++nt) {
      const int n = nBase + nt * 16 + l16;
      if (NGUARD && n >= N) continue;
      #pragma unroll
      for (int r = 0; r < 8; ++r) {
        const int m = m0 + mt * 128 + r + 8 * hi16;
        float v = acc[mt][nt][r];
        if (BIAS)   v += bias[n];
        if (GELU_)  v  = gelu_new(v);
        if (RESID)  v += resid[m * ldr + n];
        if (STF32)  Cf[m * ldc + n] = v;
        if (STBF16) Cb[m * ldcb + n] = (bf16)v;
      }
    }
  }
}

extern "C" void kernel_launch(void* const* d_in, const int* in_sizes, int n_in,
                              void* d_out, int out_size, void* d_ws, size_t ws_size,
                              hipStream_t stream) {
  (void)in_sizes; (void)n_in; (void)out_size; (void)ws_size;

  const int*   ids    = (const int*)  d_in[0];
  const float* wte    = (const float*)d_in[1];
  const float* wpe    = (const float*)d_in[2];
  const float* ln1_g  = (const float*)d_in[3];
  const float* ln1_b  = (const float*)d_in[4];
  const float* ln2_g  = (const float*)d_in[5];
  const float* ln2_b  = (const float*)d_in[6];
  const float* W_qkv  = (const float*)d_in[7];
  const float* b_qkv  = (const float*)d_in[8];
  const float* W_o    = (const float*)d_in[9];
  const float* b_o    = (const float*)d_in[10];
  const float* W_fc   = (const float*)d_in[11];
  const float* b_fc   = (const float*)d_in[12];
  const float* W_proj = (const float*)d_in[13];
  const float* b_proj = (const float*)d_in[14];
  const float* lnf_g  = (const float*)d_in[15];
  const float* lnf_b  = (const float*)d_in[16];

  // workspace carve (~46 MB total)
  char* w = (char*)d_ws;
  size_t o = 0;
  auto carve = [&](size_t bytes) -> char* {
    char* p = w + o;
    o = (o + bytes + 255) & ~(size_t)255;
    return p;
  };
  float* xA  = (float*)carve((size_t)SEQ * DM * 4);       // residual (ping)
  float* xB  = (float*)carve((size_t)SEQ * DM * 4);       // residual (pong)
  bf16*  xln = (bf16*) carve((size_t)SEQ * DM * 2);       // LN output (reused)
  bf16*  qkv = (bf16*) carve((size_t)SEQ * 3 * DM * 2);   // fused QKV
  bf16*  sc  = (bf16*) carve((size_t)NH * SEQ * SEQ * 2); // scores -> attn (in place)
  bf16*  ao  = (bf16*) carve((size_t)SEQ * DM * 2);       // attention output
  bf16*  hfc = (bf16*) carve((size_t)SEQ * FF * 2);       // gelu(fc) output

  embed_kernel<<<SEQ, 256, 0, stream>>>(ids, wte, wpe, xA);

  for (int i = 0; i < NL; ++i) {
    const float* Wqkv = W_qkv  + (size_t)i * DM * 3 * DM;
    const float* bqkv = b_qkv  + (size_t)i * 3 * DM;
    const float* Wo   = W_o    + (size_t)i * DM * DM;
    const float* bo   = b_o    + (size_t)i * DM;
    const float* Wfc  = W_fc   + (size_t)i * DM * FF;
    const float* bfc  = b_fc   + (size_t)i * FF;
    const float* Wpr  = W_proj + (size_t)i * FF * DM;
    const float* bpr  = b_proj + (size_t)i * DM;

    ln_kernel<<<SEQ, 256, 0, stream>>>(xA, ln1_g + i * DM, ln1_b + i * DM, xln);

    // qkv = xln @ Wqkv + bqkv   (bf16 out)  N=2304
    gemm_wmma<0, 3 * DM, 2, true, false, false, false, true>
        <<<dim3(36, SEQ / 256, 1), 256, 0, stream>>>(
        xln, DM, 0, Wqkv, 0, bqkv, nullptr, 0,
        nullptr, 0, 0, qkv, 3 * DM, 0, SEQ, 3 * DM, DM);

    // scores[h] = Q_h @ K_h^T  (== xln Wq Wk^T xln^T + bias cross-terms)
    gemm_wmma<1, 3 * DM, 2, false, false, false, false, true>
        <<<dim3(SEQ / 64, SEQ / 256, NH), 256, 0, stream>>>(
        qkv, 3 * DM, HD, qkv + DM, HD, nullptr, nullptr, 0,
        nullptr, 0, 0, sc, SEQ, SEQ * SEQ, SEQ, SEQ, HD);

    softmax_kernel<<<dim3(SEQ, NH), 256, 0, stream>>>(sc);

    // ao[:, h*64:(h+1)*64] = attn_h @ V_h  (rows past diagonal are exact zeros)
    gemm_wmma<2, 3 * DM, 1, false, false, false, false, true>
        <<<dim3(1, SEQ / 128, NH), 256, 0, stream>>>(
        sc, SEQ, SEQ * SEQ, qkv + 2 * DM, HD, nullptr, nullptr, 0,
        nullptr, 0, 0, ao, DM, HD, SEQ, HD, SEQ);

    // resid = x + ao @ W_o + b_o  (fp32 out)
    gemm_wmma<0, DM, 1, true, true, false, true, false>
        <<<dim3(DM / 64, SEQ / 128, 1), 256, 0, stream>>>(
        ao, DM, 0, Wo, 0, bo, xA, DM,
        xB, DM, 0, nullptr, 0, 0, SEQ, DM, DM);

    ln_kernel<<<SEQ, 256, 0, stream>>>(xB, ln2_g + i * DM, ln2_b + i * DM, xln);

    // hfc = gelu_new(h2 @ W_fc + b_fc)  (bf16 out)  N=3072
    gemm_wmma<0, FF, 2, true, false, true, false, true>
        <<<dim3(FF / 64, SEQ / 256, 1), 256, 0, stream>>>(
        xln, DM, 0, Wfc, 0, bfc, nullptr, 0,
        nullptr, 0, 0, hfc, FF, 0, SEQ, FF, DM);

    // x = resid + hfc @ W_proj + b_proj  (fp32 out, back to xA)
    gemm_wmma<0, DM, 1, true, true, false, true, false>
        <<<dim3(DM / 64, SEQ / 128, 1), 256, 0, stream>>>(
        hfc, FF, 0, Wpr, 0, bpr, xB, DM,
        xA, DM, 0, nullptr, 0, 0, SEQ, DM, FF);
  }

  ln_kernel<<<SEQ, 256, 0, stream>>>(xA, lnf_g, lnf_b, xln);

  // logits = xf @ wte^T -> d_out fp32 [S, V]  (N-edge guarded)
  gemm_wmma<3, DM, 2, false, false, false, true, false>
      <<<dim3((VOC + 63) / 64, SEQ / 256, 1), 256, 0, stream>>>(
      xln, DM, 0, wte, 0, nullptr, nullptr, 0,
      (float*)d_out, VOC, 0, nullptr, 0, 0, SEQ, VOC, DM);
}
